// TripleAttention_71116068487370
// MI455X (gfx1250) — matrix operand order repaired and checked by reading, not compiled, not run on hardware
//
#include <hip/hip_runtime.h>
#include <hip/hip_bf16.h>

// Problem constants (B=1)
#define N_TOK 2048
#define C_DIM 1024
#define H_HEADS 16
#define D_HEAD 64

typedef __attribute__((ext_vector_type(16))) __bf16 v16bf;
typedef __attribute__((ext_vector_type(8)))  float  v8f;
typedef __attribute__((ext_vector_type(16))) unsigned short us16;

#if defined(__HIP_DEVICE_COMPILE__) && __has_builtin(__builtin_amdgcn_global_load_async_to_lds_b128)
#define HAVE_ASYNC 1
#else
#define HAVE_ASYNC 0
#endif

#if HAVE_ASYNC
typedef int i32x4 __attribute__((vector_size(16)));
typedef __attribute__((address_space(1))) i32x4 gi32x4;   // global
typedef __attribute__((address_space(3))) i32x4 li32x4;   // LDS
#endif

// float -> bf16 (round-to-nearest-even), stored as raw u16
static __device__ __forceinline__ unsigned short f2bfu(float f) {
  unsigned u = __float_as_uint(f);
  u += 0x7fffu + ((u >> 16) & 1u);
  return (unsigned short)(u >> 16);
}

// Build a v16bf fragment from two 8-element (16B) groups (global or LDS).
static __device__ __forceinline__ v16bf load_frag(const unsigned short* p0,
                                                  const unsigned short* p1) {
  union { uint4 q[2]; us16 s; } t;
  t.q[0] = *(const uint4*)p0;
  t.q[1] = *(const uint4*)p1;
  return __builtin_bit_cast(v16bf, t.s);
}

// Copy 16B global -> LDS. Async (ASYNCcnt) when available, sync fallback.
static __device__ __forceinline__ void stage_b128(const unsigned short* g,
                                                  unsigned short* l) {
#if HAVE_ASYNC
  __builtin_amdgcn_global_load_async_to_lds_b128(
      (gi32x4*)g, (li32x4*)l, 0, 0);
#else
  *(uint4*)l = *(const uint4*)g;
#endif
}

// Wait for this wave's outstanding staging ops before the block barrier.
static __device__ __forceinline__ void wait_stage() {
#if HAVE_ASYNC
  asm volatile("s_wait_asynccnt 0" ::: "memory");
#endif
  asm volatile("s_wait_dscnt 0" ::: "memory");
}

#define WMMA_BF16(a, b, c) \
  __builtin_amdgcn_wmma_f32_16x16x32_bf16(false, (a), false, (b), (short)0, (c), false, false)

// ---------------------------------------------------------------- cast x->bf16
__global__ void cast_bf16_kernel(const float* __restrict__ x,
                                 unsigned short* __restrict__ y, int n) {
  int i = blockIdx.x * 256 + threadIdx.x;
  if (i < n) y[i] = f2bfu(x[i]);
}

// ------------------------------------------------- W (KxC f32) -> Wt (CxK bf16)
__global__ void wtrans_kernel(const float* __restrict__ W,
                              unsigned short* __restrict__ Wt) {
  int idx = blockIdx.x * 256 + threadIdx.x;   // idx = k*1024 + c
  int c = idx & (C_DIM - 1);
  int k = idx >> 10;
  Wt[(size_t)c * C_DIM + k] = f2bfu(W[idx]);
}

// ------------------------------------------------------------- WMMA GEMM
// out[n][c] = A[n][:] @ W[:][c] + bias[c] (+ resid[n][c])
// A: N x C bf16 row-major.  Wt: C x C bf16, transposed (Wt[c][k]).
// Block: 8 waves, 256 rows x 64 cols; wave = 32 rows (2 A-frags x 4 B-frags
// = 8 WMMA per K-step).  Weight tile (64x32) staged in LDS, double-buffered,
// async prefetch one K-step ahead.
// mode 0: store bf16 per-head  [h][n][d]          (q1,q2,k1,k2)
// mode 1: store bf16 per-head transposed [h][d][n] (v)
// mode 2: store f32 row-major N x C                (final output)
__global__ void gemm_wmma_kernel(const unsigned short* __restrict__ A,
                                 const unsigned short* __restrict__ Wt,
                                 const float* __restrict__ bias,
                                 const float* __restrict__ resid,
                                 unsigned short* __restrict__ out_bf,
                                 float* __restrict__ out_f, int mode) {
  __shared__ __align__(16) unsigned short wls[2][64][32];

  const int tid  = threadIdx.x;
  const int lane = tid & 31;
  const int w    = tid >> 5;
  const int rl   = lane & 15;
  const int hi   = (lane >= 16) ? 1 : 0;
  const int d0   = hi ? 8 : 0;    // A-frag K sub-offset
  const int kb   = hi ? 8 : 0;    // B-frag K sub-offset

  const int mbase = blockIdx.y * 256 + w * 32;  // 8 waves x 32 rows
  const int cbase = blockIdx.x * 64;            // 4 col tiles per block

  // Each thread stages 8 elements of the 64x32 weight tile.
  const int scol = tid >> 2, sseg = tid & 3;
  auto stageW = [&](int buf, int k0) {
    stage_b128(Wt + (size_t)(cbase + scol) * C_DIM + k0 + sseg * 8,
               &wls[buf][scol][sseg * 8]);
  };

  v8f acc[2][4] = {};
  const int r0 = mbase + rl;

  stageW(0, 0);
  for (int k0 = 0; k0 < C_DIM; k0 += 32) {
    const int buf = (k0 >> 5) & 1;
    wait_stage();
    __syncthreads();
    if (k0 + 32 < C_DIM) stageW(buf ^ 1, k0 + 32);

    const unsigned short* pa0 = A + (size_t)r0 * C_DIM + k0 + d0;
    const unsigned short* pa1 = pa0 + 16 * C_DIM;
    v16bf a0 = load_frag(pa0, pa0 + 16);
    v16bf a1 = load_frag(pa1, pa1 + 16);

    v16bf bfr[4];
#pragma unroll
    for (int t = 0; t < 4; ++t) {
      const unsigned short* pb = &wls[buf][t * 16 + rl][kb];
      bfr[t] = load_frag(pb, pb + 16);
    }
#pragma unroll
    for (int t = 0; t < 4; ++t) acc[0][t] = WMMA_BF16(a0, bfr[t], acc[0][t]);
#pragma unroll
    for (int t = 0; t < 4; ++t) acc[1][t] = WMMA_BF16(a1, bfr[t], acc[1][t]);
  }

#pragma unroll
  for (int m = 0; m < 2; ++m) {
#pragma unroll
    for (int t = 0; t < 4; ++t) {
      int col = cbase + t * 16 + rl;
      float bv = bias[col];
#pragma unroll
      for (int i = 0; i < 8; ++i) {
        int row = mbase + m * 16 + i + (hi ? 8 : 0);
        float val = acc[m][t][i] + bv;
        if (resid) val += resid[(size_t)row * C_DIM + col];
        if (mode == 0) {
          int h = col >> 6, d = col & 63;
          out_bf[((size_t)h * N_TOK + row) * D_HEAD + d] = f2bfu(val);
        } else if (mode == 1) {
          int h = col >> 6, d = col & 63;
          out_bf[((size_t)h * D_HEAD + d) * N_TOK + row] = f2bfu(val);
        } else {
          out_f[(size_t)row * C_DIM + col] = val;
        }
      }
    }
  }
}

// ------------------------------------------------------- flash triple attention
// s = (q1.k1^T) * (q2.k2^T), row-softmax, o = p @ v
// q1,q2,k1,k2: [h][n][64] bf16.  vT: [h][64][n] bf16.  o: [n][C] f32.
// K1/K2/V tiles for each 32-wide KV step staged in LDS (shared by all 8 waves),
// double-buffered with async prefetch of the next KV step.
__global__ void attn_kernel(const unsigned short* __restrict__ q1,
                            const unsigned short* __restrict__ q2,
                            const unsigned short* __restrict__ k1,
                            const unsigned short* __restrict__ k2,
                            const unsigned short* __restrict__ vT,
                            float* __restrict__ o) {
  __shared__ __align__(16) unsigned short plds[8][16 * 32];   // per-wave P
  __shared__ __align__(16) unsigned short k1ls[2][32][64];
  __shared__ __align__(16) unsigned short k2ls[2][32][64];
  __shared__ __align__(16) unsigned short vls[2][64][32];

  const int tid  = threadIdx.x;
  const int lane = tid & 31;
  const int w    = tid >> 5;
  const int rl   = lane & 15;
  const int hi   = (lane >= 16) ? 1 : 0;
  const int d0   = hi ? 8 : 0;
  const int kb   = hi ? 8 : 0;

  const int h     = blockIdx.y;
  const int mbase = blockIdx.x * 128 + w * 16;

  // Staging assignments: 32x64 K tiles (8 elems/thread), 64x32 V tile.
  const int krow = tid >> 3, kseg = tid & 7;
  const int vrow = tid >> 2, vseg = tid & 3;
  auto stageKV = [&](int buf, int n0) {
    stage_b128(k1 + ((size_t)h * N_TOK + n0 + krow) * D_HEAD + kseg * 8,
               &k1ls[buf][krow][kseg * 8]);
    stage_b128(k2 + ((size_t)h * N_TOK + n0 + krow) * D_HEAD + kseg * 8,
               &k2ls[buf][krow][kseg * 8]);
    stage_b128(vT + ((size_t)h * D_HEAD + vrow) * N_TOK + n0 + vseg * 8,
               &vls[buf][vrow][vseg * 8]);
  };

  // Load Q fragments once (K = d = 64 -> 2 chunks of 32)
  v16bf q1a[2], q2a[2];
#pragma unroll
  for (int kc = 0; kc < 2; ++kc) {
    const unsigned short* p =
        q1 + ((size_t)h * N_TOK + mbase + rl) * D_HEAD + kc * 32 + d0;
    q1a[kc] = load_frag(p, p + 16);
    p = q2 + ((size_t)h * N_TOK + mbase + rl) * D_HEAD + kc * 32 + d0;
    q2a[kc] = load_frag(p, p + 16);
  }

  v8f acc[4] = {};
  float mrow[8], lrow[8];
#pragma unroll
  for (int i = 0; i < 8; ++i) { mrow[i] = -3.0e38f; lrow[i] = 0.f; }

  stageKV(0, 0);
  for (int n0 = 0; n0 < N_TOK; n0 += 32) {
    const int buf = (n0 >> 5) & 1;
    wait_stage();
    __syncthreads();
    if (n0 + 32 < N_TOK) stageKV(buf ^ 1, n0 + 32);

    // Gather all 8 K-fragments, then burst the 8 score WMMAs.
    v16bf bk1[4], bk2[4];   // [t*2+kc]
#pragma unroll
    for (int t = 0; t < 2; ++t) {
      const unsigned short* pb1 = &k1ls[buf][t * 16 + rl][0];
      const unsigned short* pb2 = &k2ls[buf][t * 16 + rl][0];
#pragma unroll
      for (int kc = 0; kc < 2; ++kc) {
        bk1[t * 2 + kc] = load_frag(pb1 + kc * 32 + kb, pb1 + kc * 32 + kb + 16);
        bk2[t * 2 + kc] = load_frag(pb2 + kc * 32 + kb, pb2 + kc * 32 + kb + 16);
      }
    }
    v8f s1[2] = {}, s2[2] = {};
#pragma unroll
    for (int t = 0; t < 2; ++t)
#pragma unroll
      for (int kc = 0; kc < 2; ++kc) {
        s1[t] = WMMA_BF16(q1a[kc], bk1[t * 2 + kc], s1[t]);
        s2[t] = WMMA_BF16(q2a[kc], bk2[t * 2 + kc], s2[t]);
      }

    // elementwise product + online softmax (row stats live per-VGPR-index,
    // duplicated across each 16-lane half; reduce with xor 1,2,4,8)
    float ps0[8], ps1[8], newm[8];
#pragma unroll
    for (int i = 0; i < 8; ++i) {
      ps0[i] = s1[0][i] * s2[0][i];
      ps1[i] = s1[1][i] * s2[1][i];
      float tm = fmaxf(ps0[i], ps1[i]);
      tm = fmaxf(tm, __shfl_xor(tm, 1, 32));
      tm = fmaxf(tm, __shfl_xor(tm, 2, 32));
      tm = fmaxf(tm, __shfl_xor(tm, 4, 32));
      tm = fmaxf(tm, __shfl_xor(tm, 8, 32));
      newm[i] = fmaxf(mrow[i], tm);
    }
#pragma unroll
    for (int i = 0; i < 8; ++i) {
      float alpha = __expf(mrow[i] - newm[i]);
      float p0 = __expf(ps0[i] - newm[i]);
      float p1 = __expf(ps1[i] - newm[i]);
      float rs = p0 + p1;
      rs += __shfl_xor(rs, 1, 32);
      rs += __shfl_xor(rs, 2, 32);
      rs += __shfl_xor(rs, 4, 32);
      rs += __shfl_xor(rs, 8, 32);
      lrow[i] = lrow[i] * alpha + rs;
      mrow[i] = newm[i];
      acc[0][i] *= alpha; acc[1][i] *= alpha;
      acc[2][i] *= alpha; acc[3][i] *= alpha;
      int row = i + (hi ? 8 : 0);
      plds[w][row * 32 + rl]      = f2bfu(p0);
      plds[w][row * 32 + 16 + rl] = f2bfu(p1);
    }

    // P (16x32) as A-fragment from this wave's LDS region (DS ops are
    // in-order within a wave; no block barrier needed), then P @ V.
    const unsigned short* pp = &plds[w][rl * 32 + d0];
    v16bf pf = load_frag(pp, pp + 16);
    v16bf bv[4];
#pragma unroll
    for (int tt = 0; tt < 4; ++tt) {
      const unsigned short* pv = &vls[buf][tt * 16 + rl][kb];
      bv[tt] = load_frag(pv, pv + 16);
    }
#pragma unroll
    for (int tt = 0; tt < 4; ++tt)
      acc[tt] = WMMA_BF16(pf, bv[tt], acc[tt]);
  }

#pragma unroll
  for (int tt = 0; tt < 4; ++tt) {
    int c = h * D_HEAD + tt * 16 + rl;
#pragma unroll
    for (int i = 0; i < 8; ++i) {
      int row = mbase + i + (hi ? 8 : 0);
      o[(size_t)row * C_DIM + c] = acc[tt][i] / lrow[i];
    }
  }
}

// --------------------------------------------------------------- LayerNorm
__global__ void ln_kernel(const float* __restrict__ o,
                          const float* __restrict__ g,
                          const float* __restrict__ b,
                          unsigned short* __restrict__ oln) {
  __shared__ float sm[256], sv[256];
  int row = blockIdx.x, tid = threadIdx.x;
  float vals[4], ls = 0.f, lq = 0.f;
#pragma unroll
  for (int j = 0; j < 4; ++j) {
    int c = tid + j * 256;
    float v = o[(size_t)row * C_DIM + c];
    vals[j] = v; ls += v; lq += v * v;
  }
  sm[tid] = ls; sv[tid] = lq;
  __syncthreads();
  for (int s = 128; s > 0; s >>= 1) {
    if (tid < s) { sm[tid] += sm[tid + s]; sv[tid] += sv[tid + s]; }
    __syncthreads();
  }
  float mu = sm[0] * (1.f / C_DIM);
  float var = sv[0] * (1.f / C_DIM) - mu * mu;
  float rstd = rsqrtf(var + 1e-5f);
#pragma unroll
  for (int j = 0; j < 4; ++j) {
    int c = tid + j * 256;
    float v = (vals[j] - mu) * rstd * g[c] + b[c];
    oln[(size_t)row * C_DIM + c] = f2bfu(v);
  }
}

extern "C" void kernel_launch(void* const* d_in, const int* in_sizes, int n_in,
                              void* d_out, int out_size, void* d_ws, size_t ws_size,
                              hipStream_t stream) {
  (void)in_sizes; (void)n_in; (void)out_size; (void)ws_size;
  const float* x = (const float*)d_in[0];
  const float* W[6];  const float* bv[6];
  for (int i = 0; i < 6; ++i) {   // q1, q2, k1, k2, v, out
    W[i]  = (const float*)d_in[1 + 2 * i];
    bv[i] = (const float*)d_in[2 + 2 * i];
  }
  const float* ln_g = (const float*)d_in[13];
  const float* ln_b = (const float*)d_in[14];

  char* ws = (char*)d_ws;
  const size_t MB = 1 << 20;
  unsigned short* xbf = (unsigned short*)(ws);                 // 4 MB
  unsigned short* Wt[6];
  for (int i = 0; i < 6; ++i)
    Wt[i] = (unsigned short*)(ws + 4 * MB + (size_t)i * 2 * MB); // 12 MB
  unsigned short* q1b = (unsigned short*)(ws + 16 * MB);       // 4 MB each
  unsigned short* q2b = (unsigned short*)(ws + 20 * MB);
  unsigned short* k1b = (unsigned short*)(ws + 24 * MB);
  unsigned short* k2b = (unsigned short*)(ws + 28 * MB);
  unsigned short* vTb = (unsigned short*)(ws + 32 * MB);
  float*          obf = (float*)(ws + 36 * MB);                // 8 MB
  unsigned short* oln = (unsigned short*)(ws + 44 * MB);       // 4 MB

  const int NE = N_TOK * C_DIM;                 // 2 M elements
  cast_bf16_kernel<<<NE / 256, 256, 0, stream>>>(x, xbf, NE);
  for (int i = 0; i < 6; ++i)
    wtrans_kernel<<<(C_DIM * C_DIM) / 256, 256, 0, stream>>>(W[i], Wt[i]);

  dim3 gg(C_DIM / 64, N_TOK / 256);             // (16,8), 256 thr
  gemm_wmma_kernel<<<gg, 256, 0, stream>>>(xbf, Wt[0], bv[0], x, q1b, nullptr, 0);
  gemm_wmma_kernel<<<gg, 256, 0, stream>>>(xbf, Wt[1], bv[1], x, q2b, nullptr, 0);
  gemm_wmma_kernel<<<gg, 256, 0, stream>>>(xbf, Wt[2], bv[2], x, k1b, nullptr, 0);
  gemm_wmma_kernel<<<gg, 256, 0, stream>>>(xbf, Wt[3], bv[3], x, k2b, nullptr, 0);
  gemm_wmma_kernel<<<gg, 256, 0, stream>>>(xbf, Wt[4], bv[4], x, vTb, nullptr, 1);

  attn_kernel<<<dim3(N_TOK / 128, H_HEADS), 256, 0, stream>>>(
      q1b, q2b, k1b, k2b, vTb, obf);

  ln_kernel<<<N_TOK, 256, 0, stream>>>(obf, ln_g, ln_b, oln);

  gemm_wmma_kernel<<<gg, 256, 0, stream>>>(oln, Wt[5], bv[5], nullptr, nullptr,
                                           (float*)d_out, 2);
}
